// TGNMemory_30451318129194
// MI455X (gfx1250) — matrix operand order repaired and checked by reading, not compiled, not run on hardware
//
#include <hip/hip_runtime.h>
#include <hip/hip_bf16.h>

typedef __attribute__((ext_vector_type(16))) _Float16 v16h;
typedef __attribute__((ext_vector_type(8)))  _Float16 v8h;
typedef __attribute__((ext_vector_type(8)))  float    v8f;

#define MEMD 100
#define RAWD 172
#define TDD  100
#define MSGD 472          // MEM + MEM + RAW + TD
#define KPAD_I 480        // 472 padded to 15*32
#define KPAD_H 128        // 100 padded to 4*32
#define NT_OUT 19         // 300 cols -> 19 tiles of 16 (304 padded)
#define NT_RZ  13         // tiles covering cols 0..207 (r and z gates)
#define KT_I 15
#define KT_H 4
#define TROWS 32          // rows per block tile (2 x 16-row WMMA subtiles)
#define A_STRIDE 480      // f16 row stride, 960B = 16B aligned
#define AH_STRIDE 128
#define RZ_STRIDE 104
#define FRAG_H 512        // f16 per 32x16 B fragment

__device__ __forceinline__ float sigmoidf_(float x) {
    return 1.0f / (1.0f + __expf(-x));
}

// ------------------------------------------------------------- zero the keys
__global__ void tgn_zero_keys(unsigned long long* key, int n) {
    int i = blockIdx.x * blockDim.x + threadIdx.x;
    if (i < n) key[i] = 0ull;
}

// ----------------------------------------------- per-event argmax-(t, eidx)
__global__ void tgn_scatter(const int* __restrict__ loc_s, const int* __restrict__ t_s,
                            const int* __restrict__ loc_d, const int* __restrict__ t_d,
                            unsigned long long* __restrict__ key, int E) {
    int e = blockIdx.x * blockDim.x + threadIdx.x;
    if (e >= 2 * E) return;
    int loc, t;
    if (e < E) { loc = loc_s[e]; t = t_s[e]; }
    else       { loc = loc_d[e - E]; t = t_d[e - E]; }
    unsigned long long k = ((unsigned long long)(unsigned)(t + 1) << 32) | (unsigned)e;
    atomicMax(&key[loc], k);
}

// --------------------------- pack W_ih / W_hh into f16 WMMA B-fragment form
// B fragment (32x16, f16): lane L holds column n = L&15, slot s -> k = (L>>4)*8 + (s&7) + (s>>3)*16
// (16 halfwords per lane). One fragment = 512 f16.
__global__ void tgn_pack_w(const float* __restrict__ W_ih, const float* __restrict__ W_hh,
                           _Float16* __restrict__ wih_pack, _Float16* __restrict__ whh_pack) {
    const int TOT_I = NT_OUT * KT_I * FRAG_H;   // 145920
    const int TOT_H = NT_OUT * KT_H * FRAG_H;   //  38912
    int g = blockIdx.x * blockDim.x + threadIdx.x;
    if (g >= TOT_I + TOT_H) return;
    if (g < TOT_I) {
        int f = g / FRAG_H, r = g - f * FRAG_H;
        int lane = r >> 4, slot = r & 15;
        int nt = f / KT_I, kt = f - nt * KT_I;
        int k = kt * 32 + (lane >> 4) * 8 + (slot & 7) + (slot >> 3) * 16;
        int o = nt * 16 + (lane & 15);
        float v = (k < MSGD && o < 300) ? W_ih[(size_t)o * MSGD + k] : 0.0f;
        wih_pack[g] = (_Float16)v;
    } else {
        int g2 = g - TOT_I;
        int f = g2 / FRAG_H, r = g2 - f * FRAG_H;
        int lane = r >> 4, slot = r & 15;
        int nt = f / KT_H, kt = f - nt * KT_H;
        int k = kt * 32 + (lane >> 4) * 8 + (slot & 7) + (slot >> 3) * 16;
        int o = nt * 16 + (lane & 15);
        float v = (k < MEMD && o < 300) ? W_hh[(size_t)o * MEMD + k] : 0.0f;
        whh_pack[g2] = (_Float16)v;
    }
}

// ------------------------------------------------------------- main kernel
__global__ __launch_bounds__(128)
void tgn_main(const float* __restrict__ memory, const int* __restrict__ last_update,
              const int* __restrict__ n_id,
              const int* __restrict__ dst_s, const float* __restrict__ raw_s,
              const int* __restrict__ dst_d, const float* __restrict__ raw_d,
              const float* __restrict__ time_w, const float* __restrict__ time_b,
              const float* __restrict__ b_ih, const float* __restrict__ b_hh,
              const unsigned long long* __restrict__ key,
              const _Float16* __restrict__ wih_pack, const _Float16* __restrict__ whh_pack,
              float* __restrict__ out_mem, float* __restrict__ out_lu,
              int N, int E, int n_tiles) {
    __shared__ __align__(16) _Float16 sA [TROWS * A_STRIDE];   // 30720 B
    __shared__ __align__(16) _Float16 sAh[TROWS * AH_STRIDE];  //  8192 B
    __shared__ __align__(16) _Float16 sR [TROWS * RZ_STRIDE];  //  6656 B
    __shared__ __align__(16) _Float16 sZ [TROWS * RZ_STRIDE];  //  6656 B
    __shared__ int   sValid[TROWS], sDst[TROWS], sEid[TROWS], sGid[TROWS];
    __shared__ float sTrel[TROWS];
    __shared__ __align__(16) float sTW[100];
    __shared__ __align__(16) float sTB[100];

    const int tid   = threadIdx.x;
    const int wave  = tid >> 5;
    const int lane  = tid & 31;
    const int mrow  = lane & 15;
    const int lhalf = lane >> 4;

    // ---- async copy of time-encoder params into LDS (reused every tile) ----
    if (tid < 25) {
        unsigned ldsoff = (unsigned)(size_t)(&sTW[0]) + (unsigned)tid * 16u;
        unsigned voff = (unsigned)tid * 16u;
        asm volatile("global_load_async_to_lds_b128 %0, %1, %2"
                     :: "v"(ldsoff), "v"(voff), "s"(time_w) : "memory");
    } else if (tid >= 32 && tid < 57) {
        int t2 = tid - 32;
        unsigned ldsoff = (unsigned)(size_t)(&sTB[0]) + (unsigned)t2 * 16u;
        unsigned voff = (unsigned)t2 * 16u;
        asm volatile("global_load_async_to_lds_b128 %0, %1, %2"
                     :: "v"(ldsoff), "v"(voff), "s"(time_b) : "memory");
    }
    asm volatile("s_wait_asynccnt 0" ::: "memory");
    __syncthreads();

    for (int tile = blockIdx.x; tile < n_tiles; tile += gridDim.x) {
        // ---------------- phase 0: winner metadata + new_last_update -------
        if (tid < TROWS) {
            int row = tile * TROWS + tid;
            int valid = 0, dst = 0, e = 0, g = 0, lu = 0;
            float trel = 0.0f;
            if (row < N) {
                g = n_id[row];
                lu = last_update[g];
                unsigned long long k = key[row];
                if (k != 0ull) {
                    valid = 1;
                    e = (int)(k & 0xffffffffull);
                    int tw = (int)(k >> 32) - 1;
                    trel = (float)(tw - lu);
                    dst = (e < E) ? dst_s[e] : dst_d[e - E];
                    lu = lu > tw ? lu : tw;
                }
                out_lu[row] = (float)lu;
            }
            sValid[tid] = valid; sDst[tid] = dst; sEid[tid] = e;
            sGid[tid] = g; sTrel[tid] = trel;
        }
        __syncthreads();

        // ---------------- phase 1: stage A tiles in LDS (f32 -> f16) -------
        for (int i = tid; i < TROWS * KPAD_I; i += 128) {
            int m = i / KPAD_I, k = i - m * KPAD_I;
            float v = 0.0f;
            if (sValid[m]) {
                if (k < MEMD)            v = memory[(size_t)sGid[m] * MEMD + k];
                else if (k < 2 * MEMD)   v = memory[(size_t)sDst[m] * MEMD + (k - MEMD)];
                else if (k < 2 * MEMD + RAWD) {
                    int e = sEid[m], kk = k - 2 * MEMD;
                    v = (e < E) ? raw_s[(size_t)e * RAWD + kk]
                                : raw_d[(size_t)(e - E) * RAWD + kk];
                } else if (k < MSGD) {
                    int j = k - (2 * MEMD + RAWD);
                    v = cosf(sTrel[m] * sTW[j] + sTB[j]);
                }
            }
            sA[m * A_STRIDE + k] = (_Float16)v;
        }
        for (int i = tid; i < TROWS * KPAD_H; i += 128) {
            int m = i >> 7, k = i & 127;
            int row = tile * TROWS + m;
            float v = (k < MEMD && row < N) ? memory[(size_t)sGid[m] * MEMD + k] : 0.0f;
            sAh[m * AH_STRIDE + k] = (_Float16)v;
        }
        __syncthreads();

        // ---------- phase 2a: r/z N-tiles (cols 0..207), B reused 2x -------
        for (int nt = wave; nt < NT_RZ; nt += 4) {
            v8f a0i = {}, a1i = {}, a0h = {}, a1h = {};
            #pragma unroll
            for (int kt = 0; kt < KT_I; ++kt) {
                int k0 = kt * 32 + lhalf * 8;
                v16h b = *(const v16h*)&wih_pack[((size_t)(nt * KT_I + kt)) * FRAG_H + lane * 16];
                v8h lo0 = *(const v8h*)&sA[mrow * A_STRIDE + k0];
                v8h hi0 = *(const v8h*)&sA[mrow * A_STRIDE + k0 + 16];
                v16h a0 = __builtin_shufflevector(lo0, hi0, 0,1,2,3,4,5,6,7,8,9,10,11,12,13,14,15);
                a0i = __builtin_amdgcn_wmma_f32_16x16x32_f16(false, a0, false, b, (short)0, a0i, false, false);
                v8h lo1 = *(const v8h*)&sA[(16 + mrow) * A_STRIDE + k0];
                v8h hi1 = *(const v8h*)&sA[(16 + mrow) * A_STRIDE + k0 + 16];
                v16h a1 = __builtin_shufflevector(lo1, hi1, 0,1,2,3,4,5,6,7,8,9,10,11,12,13,14,15);
                a1i = __builtin_amdgcn_wmma_f32_16x16x32_f16(false, a1, false, b, (short)0, a1i, false, false);
            }
            #pragma unroll
            for (int kt = 0; kt < KT_H; ++kt) {
                int k0 = kt * 32 + lhalf * 8;
                v16h b = *(const v16h*)&whh_pack[((size_t)(nt * KT_H + kt)) * FRAG_H + lane * 16];
                v8h lo0 = *(const v8h*)&sAh[mrow * AH_STRIDE + k0];
                v8h hi0 = *(const v8h*)&sAh[mrow * AH_STRIDE + k0 + 16];
                v16h a0 = __builtin_shufflevector(lo0, hi0, 0,1,2,3,4,5,6,7,8,9,10,11,12,13,14,15);
                a0h = __builtin_amdgcn_wmma_f32_16x16x32_f16(false, a0, false, b, (short)0, a0h, false, false);
                v8h lo1 = *(const v8h*)&sAh[(16 + mrow) * AH_STRIDE + k0];
                v8h hi1 = *(const v8h*)&sAh[(16 + mrow) * AH_STRIDE + k0 + 16];
                v16h a1 = __builtin_shufflevector(lo1, hi1, 0,1,2,3,4,5,6,7,8,9,10,11,12,13,14,15);
                a1h = __builtin_amdgcn_wmma_f32_16x16x32_f16(false, a1, false, b, (short)0, a1h, false, false);
            }
            int c = nt * 16 + (lane & 15);
            float bi = (c < 300) ? b_ih[c] : 0.0f;
            float bh = (c < 300) ? b_hh[c] : 0.0f;
            #pragma unroll
            for (int v = 0; v < 8; ++v) {
                int m = v + 8 * lhalf;
                float u0 = (a0i[v] + bi) + (a0h[v] + bh);
                float u1 = (a1i[v] + bi) + (a1h[v] + bh);
                if (c < 100) {
                    sR[m * RZ_STRIDE + c]        = (_Float16)sigmoidf_(u0);
                    sR[(16 + m) * RZ_STRIDE + c] = (_Float16)sigmoidf_(u1);
                } else if (c < 200) {
                    sZ[m * RZ_STRIDE + (c - 100)]        = (_Float16)sigmoidf_(u0);
                    sZ[(16 + m) * RZ_STRIDE + (c - 100)] = (_Float16)sigmoidf_(u1);
                }
            }
        }
        __syncthreads();

        // ------ phase 2b: n-gate N-tiles (cols 192..303) + fused epilogue --
        for (int nt = NT_RZ - 1 + wave; nt < NT_OUT; nt += 4) {
            v8f a0i = {}, a1i = {}, a0h = {}, a1h = {};
            #pragma unroll
            for (int kt = 0; kt < KT_I; ++kt) {
                int k0 = kt * 32 + lhalf * 8;
                v16h b = *(const v16h*)&wih_pack[((size_t)(nt * KT_I + kt)) * FRAG_H + lane * 16];
                v8h lo0 = *(const v8h*)&sA[mrow * A_STRIDE + k0];
                v8h hi0 = *(const v8h*)&sA[mrow * A_STRIDE + k0 + 16];
                v16h a0 = __builtin_shufflevector(lo0, hi0, 0,1,2,3,4,5,6,7,8,9,10,11,12,13,14,15);
                a0i = __builtin_amdgcn_wmma_f32_16x16x32_f16(false, a0, false, b, (short)0, a0i, false, false);
                v8h lo1 = *(const v8h*)&sA[(16 + mrow) * A_STRIDE + k0];
                v8h hi1 = *(const v8h*)&sA[(16 + mrow) * A_STRIDE + k0 + 16];
                v16h a1 = __builtin_shufflevector(lo1, hi1, 0,1,2,3,4,5,6,7,8,9,10,11,12,13,14,15);
                a1i = __builtin_amdgcn_wmma_f32_16x16x32_f16(false, a1, false, b, (short)0, a1i, false, false);
            }
            #pragma unroll
            for (int kt = 0; kt < KT_H; ++kt) {
                int k0 = kt * 32 + lhalf * 8;
                v16h b = *(const v16h*)&whh_pack[((size_t)(nt * KT_H + kt)) * FRAG_H + lane * 16];
                v8h lo0 = *(const v8h*)&sAh[mrow * AH_STRIDE + k0];
                v8h hi0 = *(const v8h*)&sAh[mrow * AH_STRIDE + k0 + 16];
                v16h a0 = __builtin_shufflevector(lo0, hi0, 0,1,2,3,4,5,6,7,8,9,10,11,12,13,14,15);
                a0h = __builtin_amdgcn_wmma_f32_16x16x32_f16(false, a0, false, b, (short)0, a0h, false, false);
                v8h lo1 = *(const v8h*)&sAh[(16 + mrow) * AH_STRIDE + k0];
                v8h hi1 = *(const v8h*)&sAh[(16 + mrow) * AH_STRIDE + k0 + 16];
                v16h a1 = __builtin_shufflevector(lo1, hi1, 0,1,2,3,4,5,6,7,8,9,10,11,12,13,14,15);
                a1h = __builtin_amdgcn_wmma_f32_16x16x32_f16(false, a1, false, b, (short)0, a1h, false, false);
            }
            int c = nt * 16 + (lane & 15);
            if (c >= 200 && c < 300) {
                int j = c - 200;
                float bi = b_ih[c], bh = b_hh[c];
                #pragma unroll
                for (int v = 0; v < 8; ++v) {
                    int m = v + 8 * lhalf;
                    // subtile 0
                    {
                        int row = tile * TROWS + m;
                        if (row < N) {
                            float r = (float)sR[m * RZ_STRIDE + j];
                            float z = (float)sZ[m * RZ_STRIDE + j];
                            float gn = tanhf((a0i[v] + bi) + r * (a0h[v] + bh));
                            float h  = memory[(size_t)sGid[m] * MEMD + j];
                            out_mem[(size_t)row * MEMD + j] = (1.0f - z) * gn + z * h;
                        }
                    }
                    // subtile 1
                    {
                        int m1 = 16 + m;
                        int row = tile * TROWS + m1;
                        if (row < N) {
                            float r = (float)sR[m1 * RZ_STRIDE + j];
                            float z = (float)sZ[m1 * RZ_STRIDE + j];
                            float gn = tanhf((a1i[v] + bi) + r * (a1h[v] + bh));
                            float h  = memory[(size_t)sGid[m1] * MEMD + j];
                            out_mem[(size_t)row * MEMD + j] = (1.0f - z) * gn + z * h;
                        }
                    }
                }
            }
        }
        __syncthreads();   // LDS reused next tile
    }
}

// ------------------------------------------------------------------- launch
extern "C" void kernel_launch(void* const* d_in, const int* in_sizes, int n_in,
                              void* d_out, int out_size, void* d_ws, size_t ws_size,
                              hipStream_t stream) {
    const float* memory      = (const float*)d_in[0];
    const int*   last_update = (const int*)  d_in[1];
    const int*   n_id        = (const int*)  d_in[2];
    const int*   loc_s       = (const int*)  d_in[3];
    const int*   dst_s       = (const int*)  d_in[4];
    const int*   t_s         = (const int*)  d_in[5];
    const float* raw_s       = (const float*)d_in[6];
    const int*   loc_d       = (const int*)  d_in[7];
    const int*   dst_d       = (const int*)  d_in[8];
    const int*   t_d         = (const int*)  d_in[9];
    const float* raw_d       = (const float*)d_in[10];
    const float* time_w      = (const float*)d_in[11];
    const float* time_b      = (const float*)d_in[12];
    const float* W_ih        = (const float*)d_in[13];
    const float* W_hh        = (const float*)d_in[14];
    const float* b_ih        = (const float*)d_in[15];
    const float* b_hh        = (const float*)d_in[16];

    const int N = in_sizes[2];           // 200000
    const int E = in_sizes[3];           // 100000
    const int n_tiles = (N + TROWS - 1) / TROWS;   // 6250

    unsigned long long* key = (unsigned long long*)d_ws;
    const size_t key_bytes = (size_t)N * 8;                       // 1.6 MB
    _Float16* wih_pack = (_Float16*)((char*)d_ws + key_bytes);    // 285 frags
    _Float16* whh_pack = wih_pack + (size_t)NT_OUT * KT_I * FRAG_H;

    float* out_mem = (float*)d_out;
    float* out_lu  = out_mem + (size_t)N * MEMD;

    tgn_zero_keys<<<(N + 255) / 256, 256, 0, stream>>>(key, N);
    tgn_scatter<<<(2 * E + 255) / 256, 256, 0, stream>>>(loc_s, t_s, loc_d, t_d, key, E);

    const int pack_elems = NT_OUT * (KT_I + KT_H) * FRAG_H;       // 184832
    tgn_pack_w<<<(pack_elems + 255) / 256, 256, 0, stream>>>(W_ih, W_hh, wih_pack, whh_pack);

    tgn_main<<<1600, 128, 0, stream>>>(memory, last_update, n_id,
                                       dst_s, raw_s, dst_d, raw_d,
                                       time_w, time_b, b_ih, b_hh,
                                       key, wih_pack, whh_pack,
                                       out_mem, out_lu, N, E, n_tiles);
}
